// VectorQuantizer_65377992180177
// MI455X (gfx1250) — matrix operand order repaired and checked by reading, not compiled, not run on hardware
//
#include <hip/hip_runtime.h>
#include <math.h>

typedef __attribute__((ext_vector_type(16))) _Float16 v16h;
typedef __attribute__((ext_vector_type(8)))  float    v8f;
typedef __attribute__((ext_vector_type(4)))  int      v4i;

#define Bb 16
#define Dd 256
#define Tt 4096
#define NROWS (Bb*Tt)          // 65536
#define NC 1024                // codes
#define NT (NC/16)             // 64 code tiles
#define KT (Dd/32)             // 8 k tiles
#define TILE_BYTES 8192        // one code tile of B fragments (8*32*16 f16)

// workspace layout (bytes)
#define HIST_OFF   0           // uint[1024]
#define LPART_OFF  4096        // float[512]   per-block loss partials
#define ENORM_OFF  6144        // float[1024]  ||e||^2
#define EMBT_OFF   10240       // _Float16[1024*256] swizzled B fragments (512KB)

#if defined(__has_builtin)
# if __has_builtin(__builtin_amdgcn_global_load_async_to_lds_b128)
#  define HAVE_ASYNC_LDS 1
# endif
#endif

#if HAVE_ASYNC_LDS
static __device__ inline void wait_async0() {
# if __has_builtin(__builtin_amdgcn_s_wait_asynccnt)
    __builtin_amdgcn_s_wait_asynccnt(0);
# else
    asm volatile("s_wait_asynccnt 0x0" ::: "memory");
# endif
}

static __device__ inline void async_cp16(const char* g, char* s) {
    // prototype (per hipcc diagnostic): param type is int4* (generic);
    // backend infers global/LDS address spaces from the pointer operands.
    __builtin_amdgcn_global_load_async_to_lds_b128((v4i*)g, (v4i*)s, 0, 0);
}
#endif

// ---------------- prep kernels ----------------

__global__ void vq_zero(unsigned char* ws) {
    int i = blockIdx.x * blockDim.x + threadIdx.x;
    if (i < 1024) ((unsigned int*)(ws + HIST_OFF))[i] = 0u;
    if (i < 512)  ((float*)(ws + LPART_OFF))[i] = 0.0f;
}

__global__ void vq_enorm(const float* __restrict__ emb, unsigned char* ws) {
    int c = blockIdx.x * blockDim.x + threadIdx.x;   // 1024 threads
    if (c < NC) {
        float s = 0.0f;
        #pragma unroll 8
        for (int d = 0; d < Dd; ++d) { float v = emb[c * Dd + d]; s += v * v; }
        ((float*)(ws + ENORM_OFF))[c] = s;
    }
}

// Build fp16 codebook pre-swizzled into WMMA B-fragment layout:
// fragment (nt,kt): lane n (0-15) holds K = kt*32 + 0..15 for col nt*16+n,
//                   lane n (16-31) holds K = kt*32 + 16..31 for col nt*16+(n-16).
__global__ void vq_swz(const float* __restrict__ emb, unsigned char* ws) {
    int id = blockIdx.x * blockDim.x + threadIdx.x;  // 262144 total
    int i    = id & 15;
    int lane = (id >> 4) & 31;
    int kt   = (id >> 9) & 7;
    int nt   = id >> 12;
    int n = nt * 16 + (lane & 15);
    int k = kt * 32 + ((lane < 16) ? 0 : 16) + i;
    ((_Float16*)(ws + EMBT_OFF))[id] = (_Float16)emb[n * Dd + k];
}

// ---------------- main VQ kernel ----------------
// 256 threads = 8 waves; each wave owns 32 consecutive t-rows.
// B tiles (8KB each) are double-buffered in LDS and shared by all 8 waves.

__global__ void __launch_bounds__(256) vq_main(const float* __restrict__ z,
                                               const float* __restrict__ emb,
                                               float* __restrict__ out,
                                               unsigned char* __restrict__ ws) {
    __shared__ float en_sh[NC];
    __shared__ __align__(32) _Float16 btile[2][TILE_BYTES / 2];
    __shared__ int   idx_sh[8][32];
    __shared__ float wloss[8];

    const int tid  = threadIdx.x;
    const int lane = tid & 31;
    const int wv   = tid >> 5;

    const float* enorm = (const float*)(ws + ENORM_OFF);
    for (int i = tid; i < NC; i += 256) en_sh[i] = enorm[i];

    const char* embT = (const char*)(ws + EMBT_OFF);

    // stage tile 0
#if HAVE_ASYNC_LDS
    #pragma unroll
    for (int j = 0; j < 2; ++j) {
        int off = (j * 256 + tid) * 16;
        async_cp16(embT + off, (char*)&btile[0][0] + off);
    }
    wait_async0();
#else
    {
        const int4* gs = (const int4*)embT;
        int4* sd = (int4*)&btile[0][0];
        #pragma unroll
        for (int j = 0; j < 2; ++j) sd[j * 256 + tid] = gs[j * 256 + tid];
    }
#endif
    __syncthreads();

    const int row0 = (blockIdx.x * 8 + wv) * 32;     // 32 rows (t values) per wave
    const int b    = row0 / Tt;
    const int t0   = row0 % Tt;                      // same b for all 32 rows
    const float* zb = z + (size_t)b * Dd * Tt;

    const int half = lane >> 4;      // 0 or 1
    const int mrow = lane & 15;      // M (for A) / N (for B,C) within tile

    // ---- load A fragments: rows = t0 + m*16 + mrow, fp16, contiguous k order (same as B) ----
    // lanes 0-15: K = 0..15 of chunk; lanes 16-31: K = 16..31
    v16h a[2][KT];
    #pragma unroll
    for (int m = 0; m < 2; ++m) {
        #pragma unroll
        for (int kt = 0; kt < KT; ++kt) {
            v16h av;
            #pragma unroll
            for (int i = 0; i < 16; ++i) {
                int k = kt * 32 + half * 16 + i;
                av[i] = (_Float16)zb[(size_t)k * Tt + t0 + m * 16 + mrow];
            }
            a[m][kt] = av;
        }
    }

    float bestd[2][8];
    int   besti[2][8];
    #pragma unroll
    for (int m = 0; m < 2; ++m)
        #pragma unroll
        for (int v = 0; v < 8; ++v) { bestd[m][v] = 3.4e38f; besti[m][v] = 0; }

    // ---- sweep 64 code tiles, double-buffered B in LDS ----
    for (int nt = 0; nt < NT; ++nt) {
        const int cur = nt & 1;

        // prefetch next tile into the other buffer (overlaps with compute)
#if HAVE_ASYNC_LDS
        if (nt + 1 < NT) {
            const char* gt = embT + (size_t)(nt + 1) * TILE_BYTES;
            char* st = (char*)&btile[cur ^ 1][0];
            #pragma unroll
            for (int j = 0; j < 2; ++j) {
                int off = (j * 256 + tid) * 16;
                async_cp16(gt + off, st + off);
            }
        }
#else
        int4 pre[2];
        if (nt + 1 < NT) {
            const int4* gs = (const int4*)(embT + (size_t)(nt + 1) * TILE_BYTES);
            #pragma unroll
            for (int j = 0; j < 2; ++j) pre[j] = gs[j * 256 + tid];
        }
#endif

        // compute: 8 k-tiles from LDS, 2 WMMAs each
        v8f c0 = {};
        v8f c1 = {};
        const _Float16* bt = &btile[cur][0];
        #pragma unroll
        for (int kt = 0; kt < KT; ++kt) {
            v16h bf = *(const v16h*)(bt + (kt * 32 + lane) * 16);
            c0 = __builtin_amdgcn_wmma_f32_16x16x32_f16(false, a[0][kt], false, bf,
                                                        (short)0, c0, false, false);
            c1 = __builtin_amdgcn_wmma_f32_16x16x32_f16(false, a[1][kt], false, bf,
                                                        (short)0, c1, false, false);
        }
        // C layout: col N = mrow, row M = v + 8*half
        int   code = nt * 16 + mrow;
        float en   = en_sh[code];
        #pragma unroll
        for (int v = 0; v < 8; ++v) {
            float s0 = en - 2.0f * c0[v];
            if (s0 < bestd[0][v]) { bestd[0][v] = s0; besti[0][v] = code; }
            float s1 = en - 2.0f * c1[v];
            if (s1 < bestd[1][v]) { bestd[1][v] = s1; besti[1][v] = code; }
        }

#if HAVE_ASYNC_LDS
        wait_async0();
#else
        if (nt + 1 < NT) {
            int4* sd = (int4*)&btile[cur ^ 1][0];
            #pragma unroll
            for (int j = 0; j < 2; ++j) sd[j * 256 + tid] = pre[j];
        }
#endif
        __syncthreads();
    }

    // ---- per-row argmin: min-reduce across the 16 column lanes of each half ----
    unsigned int* hist = (unsigned int*)(ws + HIST_OFF);
    #pragma unroll
    for (int m = 0; m < 2; ++m) {
        #pragma unroll
        for (int v = 0; v < 8; ++v) {
            float d  = bestd[m][v];
            int   bi = besti[m][v];
            #pragma unroll
            for (int off = 8; off >= 1; off >>= 1) {
                float od = __shfl_xor(d,  off, 32);
                int   oi = __shfl_xor(bi, off, 32);
                if (od < d || (od == d && oi < bi)) { d = od; bi = oi; }
            }
            if (mrow == 0) {                         // lanes 0 and 16
                idx_sh[wv][m * 16 + half * 8 + v] = bi;
                atomicAdd(&hist[bi], 1u);
            }
        }
    }
    __syncthreads();

    // ---- output + exact f32 loss: lane <-> row (t = t0 + lane) ----
    int idx = idx_sh[wv][lane];
    const float* erow = emb + (size_t)idx * Dd;
    const size_t bbase = (size_t)b * Dd * Tt;
    float ll = 0.0f;
    #pragma unroll 4
    for (int d = 0; d < Dd; ++d) {
        float  v   = erow[d];                        // gather (codebook L2-resident)
        size_t off = (size_t)d * Tt + t0 + lane;     // coalesced across lanes
        float  zv  = zb[off];
        out[bbase + off] = v;
        float df = v - zv;
        ll += df * df;
    }
    #pragma unroll
    for (int off = 16; off >= 1; off >>= 1) ll += __shfl_xor(ll, off, 32);
    if (lane == 0) wloss[wv] = ll;
    __syncthreads();
    if (tid == 0) {
        float s = 0.0f;
        #pragma unroll
        for (int w = 0; w < 8; ++w) s += wloss[w];
        ((float*)(ws + LPART_OFF))[blockIdx.x] = s;
    }
}

// ---------------- finalize: losses + perplexity (deterministic) ----------------

__global__ void __launch_bounds__(1024) vq_fin(unsigned char* ws, float* d_out) {
    __shared__ float red[1024];
    int t = threadIdx.x;

    float lv = (t < 512) ? ((float*)(ws + LPART_OFF))[t] : 0.0f;
    red[t] = lv;
    __syncthreads();
    for (int s = 512; s > 0; s >>= 1) { if (t < s) red[t] += red[t + s]; __syncthreads(); }
    float loss = red[0];
    __syncthreads();

    unsigned int hc = ((unsigned int*)(ws + HIST_OFF))[t];
    float p = (float)hc / (float)NROWS;
    red[t] = p * logf(p + 1e-10f);
    __syncthreads();
    for (int s = 512; s > 0; s >>= 1) { if (t < s) red[t] += red[t + s]; __syncthreads(); }

    if (t == 0) {
        const size_t NO = (size_t)Bb * Dd * Tt;
        d_out[NO + 0] = loss;          // z_qut_loss
        d_out[NO + 1] = loss;          // z_enc_loss (numerically identical)
        d_out[NO + 2] = expf(-red[0]); // perplexity
    }
}

// ---------------- launcher ----------------

extern "C" void kernel_launch(void* const* d_in, const int* in_sizes, int n_in,
                              void* d_out, int out_size, void* d_ws, size_t ws_size,
                              hipStream_t stream) {
    const float* z   = (const float*)d_in[0];   // (16, 256, 4096) f32
    const float* emb = (const float*)d_in[1];   // (1024, 256) f32
    float* out = (float*)d_out;
    unsigned char* ws = (unsigned char*)d_ws;

    vq_zero <<<4,    256, 0, stream>>>(ws);
    vq_enorm<<<4,    256, 0, stream>>>(emb, ws);
    vq_swz  <<<1024, 256, 0, stream>>>(emb, ws);
    vq_main <<<256,  256, 0, stream>>>(z, emb, out, ws);
    vq_fin  <<<1,   1024, 0, stream>>>(ws, out);
}
